// SAGE_GCN_82592221102604
// MI455X (gfx1250) — matrix operand order repaired and checked by reading, not compiled, not run on hardware
//
#include <hip/hip_runtime.h>
#include <hip/hip_bf16.h>
#include <stdint.h>

#define N_NODES 20000
#define N_EDGES 320000
#define D 512
#define BM 32
#define LN_EPS 1e-5f

// LDS staging geometry for the weight K-slices
#define COL_STRIDE 80                       // 64B data + 16B pad (16B aligned, spreads banks)
#define MAT_BYTES  (512 * COL_STRIDE)       // 40960 B per matrix slice
#define STAGE_BYTES (2 * MAT_BYTES)         // Wl + Wr = 81920 B
#define SMEM_BYTES (2 * STAGE_BYTES)        // double buffered = 163840 B (2 blocks/WGP)

typedef __attribute__((ext_vector_type(16))) __bf16 v16bf;
typedef __attribute__((ext_vector_type(8)))  float  v8f;
typedef int vi4 __attribute__((vector_size(16)));   // matches async-LDS builtin pointee

union BFrag { v16bf v; uint4 u[2]; };

#if defined(__has_builtin)
#if __has_builtin(__builtin_amdgcn_global_load_async_to_lds_b128)
#define HAVE_ASYNC_COPY 1
#endif
#endif

#if defined(__has_builtin)
#if __has_builtin(__builtin_amdgcn_s_wait_asynccnt)
#define WAIT_ASYNC(n) __builtin_amdgcn_s_wait_asynccnt(n)
#endif
#endif
#ifndef WAIT_ASYNC
#define WAIT_ASYNC(n) asm volatile("s_wait_asynccnt %0" ::"i"(n) : "memory")
#endif

// ---------------------------------------------------------------------------
// Degree count + reciprocal (layer invariant)
// ---------------------------------------------------------------------------
__global__ __launch_bounds__(256) void k_count(const long long* __restrict__ ei,
                                               float* __restrict__ cnt, int nE) {
    int e = blockIdx.x * 256 + threadIdx.x;
    if (e < nE) atomicAdd(&cnt[(int)ei[nE + e]], 1.0f);
}

__global__ __launch_bounds__(256) void k_inv(float* __restrict__ cnt, int n) {
    int i = blockIdx.x * 256 + threadIdx.x;
    if (i < n) cnt[i] = 1.0f / fmaxf(cnt[i], 1.0f);
}

// ---------------------------------------------------------------------------
// Edge scatter-add: each thread handles 4 consecutive features of one edge
// ---------------------------------------------------------------------------
__global__ __launch_bounds__(256) void k_scatter(const long long* __restrict__ ei,
                                                 const float* __restrict__ h,
                                                 float* __restrict__ acc, int nE) {
    long long tid = (long long)blockIdx.x * 256 + threadIdx.x;
    int e = (int)(tid >> 7);
    int f = (int)(tid & 127) << 2;
    if (e < nE) {
        long long s = ei[e];
        long long d = ei[nE + e];
        const float* hp = h + s * D + f;
        float* ap = acc + d * D + f;
        atomicAdd(ap + 0, hp[0]);
        atomicAdd(ap + 1, hp[1]);
        atomicAdd(ap + 2, hp[2]);
        atomicAdd(ap + 3, hp[3]);
    }
}

__global__ __launch_bounds__(256) void k_finalize(const float* __restrict__ acc,
                                                  const float* __restrict__ inv,
                                                  __bf16* __restrict__ aggb, int n) {
    int i = blockIdx.x * 256 + threadIdx.x;
    if (i < n) aggb[i] = (__bf16)(acc[i] * inv[i >> 9]);
}

__global__ __launch_bounds__(256) void k_cvt(const float* __restrict__ src,
                                             __bf16* __restrict__ dst, int n) {
    int i = blockIdx.x * 256 + threadIdx.x;
    if (i < n) dst[i] = (__bf16)src[i];
}

// ---------------------------------------------------------------------------
// Fused: out = PReLU(LN(agg@Wl.T + bl + h@Wr.T + residual))
// Block = 256 threads (8 waves), tile = 32 rows x 512 cols.
// Weights double-buffered in LDS via async copies; A frags pipelined in regs;
// B fragments batched 8-wide so one s_wait_dscnt covers 8 consecutive WMMAs.
// ---------------------------------------------------------------------------
__global__ __launch_bounds__(256)
void k_sage_gemm(const __bf16* __restrict__ aggb,   // [N,512] bf16
                 const __bf16* __restrict__ hb,     // [N,512] bf16
                 const __bf16* __restrict__ wlb,    // [512,512] bf16 (row j = out col j)
                 const __bf16* __restrict__ wrb,    // [512,512] bf16
                 const float*  __restrict__ bl,
                 const float*  __restrict__ resid,
                 const float*  __restrict__ gamma,
                 const float*  __restrict__ beta,
                 const float*  __restrict__ aPtr,
                 float* __restrict__ out,
                 int useRes) {
    extern __shared__ char smem[];

    const int tid  = threadIdx.x;
    const int lane = tid & 31;
    const int wave = tid >> 5;
    const int m    = lane & 15;
    const int half = lane >> 4;
    const int r    = wave >> 2;          // row tile 0..1
    const int c    = wave & 3;           // 128-col group 0..3
    const int rowBlk = blockIdx.x * BM;
    const int row0   = rowBlk + r * 16;
    const int cbase  = c * 128;

    v8f acc[8];
#pragma unroll
    for (int t = 0; t < 8; ++t) acc[t] = v8f{};

    const __bf16* aAgg = aggb + (size_t)(row0 + m) * D;
    const __bf16* aH   = hb   + (size_t)(row0 + m) * D;

    // --- cooperative stage copy: Wl+Wr K-slice -> LDS (4096 x 16B, 16/thread) ---
    auto issueStage = [&](int stage, int kk) {
#pragma unroll
        for (int q = 0; q < 16; ++q) {
            const int idx = tid + q * 256;       // 0..4095
            const int mat = idx >> 11;           // 0=Wl 1=Wr (uniform per q)
            const int rem = idx & 2047;
            const int col = rem >> 2;
            const int seg = rem & 3;
            const __bf16* gsrc = (mat ? wrb : wlb) + (size_t)col * D + kk + seg * 8;
            const int ldsOff = stage * STAGE_BYTES + mat * MAT_BYTES +
                               col * COL_STRIDE + seg * 16;
#ifdef HAVE_ASYNC_COPY
            __builtin_amdgcn_global_load_async_to_lds_b128(
                (__attribute__((address_space(1))) vi4*)(void*)gsrc,
                (__attribute__((address_space(3))) vi4*)(smem + ldsOff),
                0, 0);
#else
            *(uint4*)(smem + ldsOff) = *(const uint4*)gsrc;
#endif
        }
    };

    auto loadA = [&](int kk, BFrag& fa, BFrag& fh) {
        fa.u[0] = *(const uint4*)(aAgg + kk + half * 8);
        fa.u[1] = *(const uint4*)(aAgg + kk + 16 + half * 8);
        fh.u[0] = *(const uint4*)(aH   + kk + half * 8);
        fh.u[1] = *(const uint4*)(aH   + kk + 16 + half * 8);
    };

    // matOff = 0 (Wl) or MAT_BYTES (Wr)
    auto loadB = [&](int stage, int t, int matOff, BFrag& b) {
        const char* p0 = smem + stage * STAGE_BYTES + matOff +
                         (cbase + t * 16 + m) * COL_STRIDE + half * 16;
        b.u[0] = *(const uint4*)(p0);
        b.u[1] = *(const uint4*)(p0 + 32);
    };

    // ---- prologue: stage 0 + first A fragments in flight ----
    issueStage(0, 0);
    BFrag fa, fh, faN, fhN;
    loadA(0, fa, fh);
#ifdef HAVE_ASYNC_COPY
    WAIT_ASYNC(0);
#endif
    __syncthreads();

    for (int it = 0; it < 16; ++it) {
        const int cur = it & 1;
        const int kk  = it * 32;
        if (it < 15) {
            issueStage(cur ^ 1, kk + 32);    // overlap next stage with compute
            loadA(kk + 32, faN, fhN);
            if (it < 14) {
                __builtin_prefetch(aAgg + kk + 64, 0, 3);
                __builtin_prefetch(aH   + kk + 64, 0, 3);
            }
        }

        BFrag bfr[8];
        // batch 1: all 8 Wl fragments, then 8 back-to-back WMMAs
#pragma unroll
        for (int t = 0; t < 8; ++t) loadB(cur, t, 0, bfr[t]);
#pragma unroll
        for (int t = 0; t < 8; ++t)
            acc[t] = __builtin_amdgcn_wmma_f32_16x16x32_bf16(
                false, fa.v, false, bfr[t].v, (short)0, acc[t], false, false);
        // batch 2: all 8 Wr fragments (reuse regs), then 8 back-to-back WMMAs
#pragma unroll
        for (int t = 0; t < 8; ++t) loadB(cur, t, MAT_BYTES, bfr[t]);
#pragma unroll
        for (int t = 0; t < 8; ++t)
            acc[t] = __builtin_amdgcn_wmma_f32_16x16x32_bf16(
                false, fh.v, false, bfr[t].v, (short)0, acc[t], false, false);

        if (it < 15) {
            fa = faN; fh = fhN;
#ifdef HAVE_ASYNC_COPY
            WAIT_ASYNC(0);
#endif
            __syncthreads();
        }
    }

    // ---- epilogue: bias + residual, LayerNorm, PReLU (reuse stage LDS) ----
    __syncthreads();
    float* s_sum = (float*)smem;             // [32][65]
    float* s_sq  = s_sum + 32 * 65;
    float* s_mu  = s_sq  + 32 * 65;
    float* s_rs  = s_mu  + 32;

    float ps[8], ps2[8];
#pragma unroll
    for (int i = 0; i < 8; ++i) { ps[i] = 0.f; ps2[i] = 0.f; }

#pragma unroll
    for (int t = 0; t < 8; ++t) {
        const int col  = cbase + t * 16 + m;
        const float blv = bl[col];
#pragma unroll
        for (int i = 0; i < 8; ++i) {
            const int rowg = row0 + i + half * 8;
            float v = acc[t][i] + blv;
            if (useRes) v += resid[(size_t)rowg * D + col];
            acc[t][i] = v;
            ps[i]  += v;
            ps2[i] += v * v;
        }
    }
#pragma unroll
    for (int i = 0; i < 8; ++i) {
        const int rb = r * 16 + i + half * 8;
        s_sum[rb * 65 + c * 16 + m] = ps[i];
        s_sq [rb * 65 + c * 16 + m] = ps2[i];
    }
    __syncthreads();

    if (tid < BM) {
        float s = 0.f, q = 0.f;
        for (int j = 0; j < 64; ++j) { s += s_sum[tid * 65 + j]; q += s_sq[tid * 65 + j]; }
        const float mu  = s * (1.0f / D);
        const float var = q * (1.0f / D) - mu * mu;
        s_mu[tid] = mu;
        s_rs[tid] = rsqrtf(var + LN_EPS);
    }
    __syncthreads();

    const float slope = aPtr[0];
#pragma unroll
    for (int t = 0; t < 8; ++t) {
        const int col = cbase + t * 16 + m;
        const float gvv = gamma[col];
        const float bvv = beta[col];
#pragma unroll
        for (int i = 0; i < 8; ++i) {
            const int rb   = r * 16 + i + half * 8;
            const int rowg = rowBlk + rb;
            float v = (acc[t][i] - s_mu[rb]) * s_rs[rb] * gvv + bvv;
            v = (v >= 0.f) ? v : slope * v;
            out[(size_t)rowg * D + col] = v;
        }
    }
}

// ---------------------------------------------------------------------------
// Host launcher
// ---------------------------------------------------------------------------
static inline size_t alignUp(size_t v, size_t a) { return (v + a - 1) & ~(a - 1); }

extern "C" void kernel_launch(void* const* d_in, const int* in_sizes, int n_in,
                              void* d_out, int out_size, void* d_ws, size_t ws_size,
                              hipStream_t stream) {
    (void)in_sizes; (void)n_in; (void)out_size; (void)ws_size;

    const float*     x  = (const float*)d_in[0];
    const long long* ei = (const long long*)d_in[1];
    const float* Wl[3] = {(const float*)d_in[2],  (const float*)d_in[7],  (const float*)d_in[12]};
    const float* blv[3]= {(const float*)d_in[3],  (const float*)d_in[8],  (const float*)d_in[13]};
    const float* Wr[3] = {(const float*)d_in[4],  (const float*)d_in[9],  (const float*)d_in[14]};
    const float* gv[3] = {(const float*)d_in[5],  (const float*)d_in[10], (const float*)d_in[15]};
    const float* bv[3] = {(const float*)d_in[6],  (const float*)d_in[11], (const float*)d_in[16]};
    const float* aP    = (const float*)d_in[17];

    const size_t nodeF32 = (size_t)N_NODES * D * sizeof(float);
    const size_t nodeBF  = (size_t)N_NODES * D * sizeof(__bf16);
    const size_t wBF     = (size_t)D * D * sizeof(__bf16);

    char* ws = (char*)d_ws;
    size_t off = 0;
    float*  cnt  = (float*)(ws + off);  off = alignUp(off + N_NODES * sizeof(float), 256);
    float*  acc  = (float*)(ws + off);  off = alignUp(off + nodeF32, 256);
    __bf16* hb   = (__bf16*)(ws + off); off = alignUp(off + nodeBF, 256);
    __bf16* aggb = (__bf16*)(ws + off); off = alignUp(off + nodeBF, 256);
    __bf16* wlb  = (__bf16*)(ws + off); off = alignUp(off + wBF, 256);
    __bf16* wrb  = (__bf16*)(ws + off); off = alignUp(off + wBF, 256);
    float*  h1   = (float*)(ws + off);  off = alignUp(off + nodeF32, 256);
    float*  h2   = (float*)(ws + off);  off = alignUp(off + nodeF32, 256);

    const int nElem = N_NODES * D;

    (void)hipMemsetAsync(cnt, 0, N_NODES * sizeof(float), stream);
    k_count<<<(N_EDGES + 255) / 256, 256, 0, stream>>>(ei, cnt, N_EDGES);
    k_inv<<<(N_NODES + 255) / 256, 256, 0, stream>>>(cnt, N_NODES);

    const float* hin = x;
    float* outs[3] = {h1, h2, (float*)d_out};

    for (int L = 0; L < 3; ++L) {
        (void)hipMemsetAsync(acc, 0, nodeF32, stream);
        k_cvt<<<(nElem + 255) / 256, 256, 0, stream>>>(hin, hb, nElem);
        k_cvt<<<(D * D + 255) / 256, 256, 0, stream>>>(Wl[L], wlb, D * D);
        k_cvt<<<(D * D + 255) / 256, 256, 0, stream>>>(Wr[L], wrb, D * D);

        const long long scatterThreads = (long long)N_EDGES * (D / 4);
        k_scatter<<<(unsigned)((scatterThreads + 255) / 256), 256, 0, stream>>>(ei, hin, acc, N_EDGES);

        k_finalize<<<(nElem + 255) / 256, 256, 0, stream>>>(acc, cnt, aggb, nElem);

        k_sage_gemm<<<N_NODES / BM, 256, SMEM_BYTES, stream>>>(
            aggb, hb, wlb, wrb, blv[L],
            (L > 0) ? hin : x, gv[L], bv[L], aP, outs[L], (L > 0) ? 1 : 0);

        hin = outs[L];
    }
}
